// GRUODE_50311246905422
// MI455X (gfx1250) — compile-verified, matrix operand order checked by reading, not run on hardware
//
#include <hip/hip_runtime.h>
#include <hip/hip_bf16.h>

typedef __attribute__((ext_vector_type(16))) _Float16 v16h;
typedef __attribute__((ext_vector_type(8)))  float    v8f;

#define B_     128
#define TIN_   512
#define NF_    33
#define CIN_   64
#define H_     256
#define COUT_  64
#define WIDTH_ 128
#define MAX_STEPS_ 16
#define MS     16            // samples per block (WMMA M tile)
#define NBLK   (B_/MS)       // 8 workgroups
#define NTHREADS 512
#define NWAVES 16

// ---- packed f16 weight fragment offsets in d_ws (units: halfs) ----
// layout per weight (N x K): [tile j][frag f][lane L][elem e], 512 halfs per (j,f)
#define SZ_WIH (48*2*512)
#define SZ_WHH (48*8*512)
#define SZ_W0  (8*8*512)
#define SZ_W1  (8*4*512)
#define SZ_W2  (16*4*512)
#define SZ_RO  (4*8*512)
#define OFF_WIH 0
#define OFF_WHH (OFF_WIH+SZ_WIH)
#define OFF_W0  (OFF_WHH+SZ_WHH)
#define OFF_W1  (OFF_W0+SZ_W0)
#define OFF_W2  (OFF_W1+SZ_W1)
#define OFF_RO  (OFF_W2+SZ_W2)
#define PACK_TOTAL (OFF_RO+SZ_RO)

// dynamic LDS layout (bytes)
#define APACK_OFF 0            // 8 frags * 512 halfs = 8KB
#define XPACK_OFF 8192         // 2 frags * 512 halfs = 2KB
#define U_OFF     10240        // union region
// GRU phase:  gates_s [MS*768] f32 (48KB) | gates_hn [MS*H] f32 (16KB) | wih_lds (96KB)
// ODE phase:  act [MS*H] f32 (16KB) | act2 [MS*WIDTH] f32 (8KB) | wlds (192KB)
#define SMEM_TOTAL (U_OFF + 16384 + 8192 + (SZ_W0+SZ_W1+SZ_W2+SZ_RO)*2)

// K-offset inside a 16x16x32 f16 fragment for (lane L, element e)
__device__ __forceinline__ int frag_koff(int L, int e) {
    return e + ((e >= 8) ? 8 : 0) + ((L >= 16) ? 8 : 0);
}

// Pack one element of W (N x K, row-major) into B-fragment order.
__device__ __forceinline__ void pack_one(const float* W, int K, int F,
                                         _Float16* dst, long idx) {
    int  e    = (int)(idx & 15);
    int  L    = (int)((idx >> 4) & 31);
    long rest = idx >> 9;
    int  f    = (int)(rest % F);
    int  j    = (int)(rest / F);
    int  n    = j * 16 + (L & 15);
    int  k    = f * 32 + frag_koff(L, e);
    dst[idx]  = (_Float16)W[(long)n * K + k];
}

__global__ void pack_weights_kernel(const float* __restrict__ wih,
                                    const float* __restrict__ whh,
                                    const float* __restrict__ w0,
                                    const float* __restrict__ w1,
                                    const float* __restrict__ w2,
                                    const float* __restrict__ ro,
                                    _Float16* __restrict__ ws) {
    long tid    = (long)blockIdx.x * blockDim.x + threadIdx.x;
    long stride = (long)gridDim.x * blockDim.x;
    for (long i = tid; i < PACK_TOTAL; i += stride) {
        if      (i < OFF_WHH) pack_one(wih,  64, 2, ws + OFF_WIH, i - OFF_WIH);
        else if (i < OFF_W0)  pack_one(whh, 256, 8, ws + OFF_WHH, i - OFF_WHH);
        else if (i < OFF_W1)  pack_one(w0,  256, 8, ws + OFF_W0,  i - OFF_W0);
        else if (i < OFF_W2)  pack_one(w1,  128, 4, ws + OFF_W1,  i - OFF_W1);
        else if (i < OFF_RO)  pack_one(w2,  128, 4, ws + OFF_W2,  i - OFF_W2);
        else                  pack_one(ro,  256, 8, ws + OFF_RO,  i - OFF_RO);
    }
}

__device__ __forceinline__ float softplusf(float x) {
    return fmaxf(x, 0.0f) + log1pf(expf(-fabsf(x)));
}
__device__ __forceinline__ float sigmoidf_(float x) {
    return 1.0f / (1.0f + expf(-x));
}

// Opaque zero: defeats LICM/CSE across loop iterations while PRESERVING pointer
// provenance (keeps global loads global and LDS loads ds_, no FLAT fallback).
__device__ __forceinline__ int opaque0() {
    int x = 0;
    asm volatile("" : "+v"(x));
    return x;
}

// Store value v of logical A element (row m, k = frag*32 + k32) into A-fragment LDS buffer.
__device__ __forceinline__ void pack_store(_Float16* dst, int frag, int m, int k32, float v) {
    int sub  = k32 & 15;
    int lane = (sub < 8) ? m : (m + 16);
    int e    = ((sub < 8) ? sub : (sub - 8)) + ((k32 >= 16) ? 8 : 0);
    dst[((frag * 32 + lane) << 4) + e] = (_Float16)v;
}

// Accumulate C += A(16 x 32*nfrag, LDS frags) * B(32*nfrag x 16, packed frags)
__device__ __forceinline__ v8f wmma_acc(const _Float16* apack, const _Float16* bp,
                                        int nfrag, int lane, v8f acc) {
    for (int f = 0; f < nfrag; ++f) {
        v16h a = *(const v16h*)(apack + ((f * 32 + lane) << 4));
        v16h b = *(const v16h*)(bp    + ((f * 32 + lane) << 4));
        acc = __builtin_amdgcn_wmma_f32_16x16x32_f16(false, a, false, b,
                                                     (short)0, acc, false, false);
    }
    return acc;
}

__global__ __launch_bounds__(NTHREADS)
void gruode_main(const float* __restrict__ y_past, const float* __restrict__ tvec,
                 const int* __restrict__ input_length,
                 const float* __restrict__ gru_b, const float* __restrict__ gru_bn,
                 const float* __restrict__ b0, const float* __restrict__ b1,
                 const float* __restrict__ b2, const float* __restrict__ ro_b,
                 const _Float16* __restrict__ wpack, float* __restrict__ out) {
    extern __shared__ char smem[];
    _Float16* apack = (_Float16*)(smem + APACK_OFF);
    _Float16* xpack = (_Float16*)(smem + XPACK_OFF);
    // GRU-phase view of union region
    float*    gates_s  = (float*)(smem + U_OFF);             // [MS*768]
    float*    gates_hn = gates_s + MS * 768;                 // [MS*H]
    _Float16* wih_lds  = (_Float16*)(smem + U_OFF + 65536);  // SZ_WIH halfs
    // ODE-phase view of union region
    float*    act  = (float*)(smem + U_OFF);                 // [MS*H]
    float*    act2 = act + MS * H_;                          // [MS*WIDTH]
    _Float16* wlds = (_Float16*)(smem + U_OFF + 24576);      // w0|w1|w2|ro frags
    _Float16* w0_l = wlds;
    _Float16* w1_l = w0_l + SZ_W0;
    _Float16* w2_l = w1_l + SZ_W1;
    _Float16* ro_l = w2_l + SZ_W2;

    __shared__ float t_cur[MS], dtv[MS], dt_use[MS];
    __shared__ int   done_s[MS];

    const int tid  = threadIdx.x;
    const int lane = tid & 31;
    const int wave = tid >> 5;     // == sample index m for elementwise work
    const int g    = blockIdx.x;

    const _Float16* whh_p = wpack + OFF_WHH;

    // ---- per-thread register state: element (m=wave, j=lane+32*jj) ----
    float zr[8];                   // GRU hidden / ODE state
    float kr[7][8];                // RK slopes
    float y5r[8];                  // 5th-order candidate
#pragma unroll
    for (int jj = 0; jj < 8; ++jj) {
        zr[jj] = 0.0f; y5r[jj] = 0.0f;
#pragma unroll
        for (int s = 0; s < 7; ++s) kr[s][jj] = 0.0f;
    }
    // init packed A fragments (h = 0)
#pragma unroll
    for (int jj = 0; jj < 8; ++jj) pack_store(apack, jj, wave, lane, 0.0f);
    // stage wih fragments into LDS (96KB, used by all 512 GRU steps)
    {
        const unsigned long long* src = (const unsigned long long*)(wpack + OFF_WIH);
        unsigned long long*       dst = (unsigned long long*)wih_lds;
        for (int i = tid; i < SZ_WIH / 4; i += NTHREADS) dst[i] = src[i];
    }
    __syncthreads();

    // ---------------- GRU over TIN steps ----------------
    for (int t = 0; t < TIN_; ++t) {
        // stage x_t into A-fragment layout
#pragma unroll
        for (int rep = 0; rep < 2; ++rep) {
            int i = tid + rep * NTHREADS;
            int m = i >> 6, c = i & 63;
            float v = y_past[(((long)(g * MS + m)) * TIN_ + t) * CIN_ + c];
            pack_store(xpack, c >> 5, m, c & 31, v);
        }
        __syncthreads();
        // gates GEMM: ig = X*wih^T, hg = H*whh^T ; 3 N-tiles per wave
        {
            int oq = opaque0();
            const _Float16* wih_o = wih_lds + oq;   // LDS, provenance kept
            const _Float16* whh_o = whh_p   + oq;   // global (L2-resident)
#pragma unroll
            for (int q = 0; q < 3; ++q) {
                int tile = wave * 3 + q;
                v8f ai = {}; v8f ah = {};
                ai = wmma_acc(xpack, wih_o + (long)tile * 2 * 512, 2, lane, ai);
                ah = wmma_acc(apack, whh_o + (long)tile * 8 * 512, 8, lane, ah);
                int n0 = tile * 16 + (lane & 15);
                int mb = (lane >> 4) * 8;
#pragma unroll
                for (int v = 0; v < 8; ++v) {
                    int mr = mb + v;
                    if (n0 < 2 * H_) {
                        gates_s[mr * 768 + n0] = ai[v] + ah[v] + gru_b[n0];
                    } else {
                        gates_s[mr * 768 + n0] = ai[v] + gru_b[n0];        // inn
                        gates_hn[mr * H_ + (n0 - 2 * H_)] = ah[v];         // hn
                    }
                }
            }
        }
        __syncthreads();
        // elementwise GRU cell update (state in registers)
        {
            int m = wave;
#pragma unroll
            for (int jj = 0; jj < 8; ++jj) {
                int j = lane + jj * 32;
                float sr = gates_s[m * 768 + j];
                float sz = gates_s[m * 768 + H_ + j];
                float sn = gates_s[m * 768 + 2 * H_ + j];
                float hn = gates_hn[m * H_ + j];
                float r  = sigmoidf_(sr);
                float zg = sigmoidf_(sz);
                float nn = tanhf(sn + r * (hn + gru_bn[j]));
                float hv = (1.0f - zg) * nn + zg * zr[jj];
                zr[jj] = hv;
                pack_store(apack, jj, m, lane, hv);
            }
        }
        __syncthreads();
    }

    // stage ODE-phase weights (w0|w1|w2|ro fragments, 192KB) into LDS
    {
        const unsigned long long* src = (const unsigned long long*)(wpack + OFF_W0);
        unsigned long long*       dst = (unsigned long long*)wlds;
        for (int i = tid; i < (SZ_W0 + SZ_W1 + SZ_W2 + SZ_RO) / 4; i += NTHREADS)
            dst[i] = src[i];
    }
    __syncthreads();

    // readout of current register state into out[:, row, :]
    auto readout = [&](int row) {
#pragma unroll
        for (int jj = 0; jj < 8; ++jj) pack_store(apack, jj, wave, lane, zr[jj]);
        __syncthreads();
        if (wave < 8) {                       // 4 tiles x 2 K-halves
            int tile = wave & 3, kh = wave >> 2;
            const _Float16* ro_o = ro_l + opaque0();
            v8f acc = {};
            acc = wmma_acc(apack + kh * 4 * 512,
                           ro_o + ((long)tile * 8 + kh * 4) * 512, 4, lane, acc);
            int n0 = tile * 16 + (lane & 15);
            int mb = (lane >> 4) * 8;
            float* dst = kh ? act2 : act;
#pragma unroll
            for (int v = 0; v < 8; ++v) dst[(mb + v) * COUT_ + n0] = acc[v];
        }
        __syncthreads();
        {
            int m = wave;
#pragma unroll
            for (int rep = 0; rep < 2; ++rep) {
                int n = lane + rep * 32;
                out[(((long)(g * MS + m)) * NF_ + row) * COUT_ + n] =
                    act[m * COUT_ + n] + act2[m * COUT_ + n] + ro_b[n];
            }
        }
        __syncthreads();
    };

    const int il = input_length[0];
    const float* tf = tvec + il;
    const float dt0 = (tf[NF_ - 1] - tf[0]) * 0.01f;

    readout(0);   // ys[0] = zc

    if (tid < MS) { t_cur[tid] = tf[0]; dtv[tid] = dt0; }
    __syncthreads();

    // ---------------- adaptive RK45 over 32 intervals ----------------
    for (int iv = 1; iv < NF_; ++iv) {
        float t_next = tf[iv];
        for (int st = 0; st < MAX_STEPS_; ++st) {
            if (tid < MS) {
                float rem   = t_next - t_cur[tid];
                done_s[tid] = (rem <= 1e-8f) ? 1 : 0;
                dt_use[tid] = fminf(fmaxf(dtv[tid], 0.0f), fmaxf(rem, 0.0f));
            }
            __syncthreads();
            const float du = dt_use[wave];

            // Dormand-Prince tableau: row s gives the input of stage s+1 (row 5 = y5)
            const float AY[6][6] = {
                {0.2f, 0, 0, 0, 0, 0},
                {3.f/40, 9.f/40, 0, 0, 0, 0},
                {44.f/45, -56.f/15, 32.f/9, 0, 0, 0},
                {19372.f/6561, -25360.f/2187, 64448.f/6561, -212.f/729, 0, 0},
                {9017.f/3168, -355.f/33, 46732.f/5247, 49.f/176, -5103.f/18656, 0},
                {35.f/384, 0, 500.f/1113, 125.f/192, -2187.f/6784, 11.f/84}
            };

            // pack stage-0 input (current z)
#pragma unroll
            for (int jj = 0; jj < 8; ++jj) pack_store(apack, jj, wave, lane, zr[jj]);
            __syncthreads();

            // 7 RK stages, fully unrolled so kr[][] stays in registers
#pragma unroll
            for (int s = 0; s < 7; ++s) {
                int oq = opaque0();
                // L0: (16x256)x(256x128), K split across wave pairs
                {
                    int tile = wave & 7, kh = wave >> 3;
                    v8f acc = {};
                    acc = wmma_acc(apack + kh * 4 * 512,
                                   w0_l + oq + ((long)tile * 8 + kh * 4) * 512, 4, lane, acc);
                    int n0 = tile * 16 + (lane & 15);
                    int mb = (lane >> 4) * 8;
                    float* dst = kh ? act2 : act;
#pragma unroll
                    for (int v = 0; v < 8; ++v) dst[(mb + v) * WIDTH_ + n0] = acc[v];
                }
                __syncthreads();
                {
                    int m = wave;
#pragma unroll
                    for (int jj = 0; jj < 4; ++jj) {
                        int j = lane + jj * 32;
                        float hv = softplusf(act[m * WIDTH_ + j] + act2[m * WIDTH_ + j] + b0[j]);
                        pack_store(apack, jj, m, lane, hv);
                    }
                }
                __syncthreads();
                // L1: (16x128)x(128x128), K split across wave pairs
                {
                    int tile = wave & 7, kh = wave >> 3;
                    v8f acc = {};
                    acc = wmma_acc(apack + kh * 2 * 512,
                                   w1_l + oq + ((long)tile * 4 + kh * 2) * 512, 2, lane, acc);
                    int n0 = tile * 16 + (lane & 15);
                    int mb = (lane >> 4) * 8;
                    float* dst = kh ? act2 : act;
#pragma unroll
                    for (int v = 0; v < 8; ++v) dst[(mb + v) * WIDTH_ + n0] = acc[v];
                }
                __syncthreads();
                {
                    int m = wave;
#pragma unroll
                    for (int jj = 0; jj < 4; ++jj) {
                        int j = lane + jj * 32;
                        float hv = softplusf(act[m * WIDTH_ + j] + act2[m * WIDTH_ + j] + b1[j]);
                        pack_store(apack, jj, m, lane, hv);
                    }
                }
                __syncthreads();
                // L2: (16x128)x(128x256), 16 tiles across 16 waves
                {
                    v8f acc = {};
                    acc = wmma_acc(apack, w2_l + oq + (long)wave * 4 * 512, 4, lane, acc);
                    int n0 = wave * 16 + (lane & 15);
                    int mb = (lane >> 4) * 8;
#pragma unroll
                    for (int v = 0; v < 8; ++v) act[(mb + v) * H_ + n0] = acc[v];
                }
                __syncthreads();
                // k_s = tanh(...) ; merged: immediately build and pack stage s+1 input
                {
                    int m = wave;
#pragma unroll
                    for (int jj = 0; jj < 8; ++jj) {
                        int j = lane + jj * 32;
                        float kv = tanhf(act[m * H_ + j] + b2[j]);
                        kr[s][jj] = kv;
                        if (s < 6) {
                            float a = 0.0f;
#pragma unroll
                            for (int p = 0; p <= s; ++p) a += AY[s][p] * kr[p][jj];
                            float zs = zr[jj] + du * a;
                            if (s == 5) y5r[jj] = zs;     // stage-7 input is y5
                            pack_store(apack, jj, m, lane, zs);
                        }
                    }
                }
                __syncthreads();
            }
            // error norm, accept/reject, step-size control (registers + wave shuffle)
            {
                int m = wave;
                const float EC0 = 71.f/57600, EC2 = -71.f/16695, EC3 = 71.f/1920,
                            EC4 = -17253.f/339200, EC5 = 22.f/525, EC6 = -1.f/40;
                float psum = 0.0f;
#pragma unroll
                for (int jj = 0; jj < 8; ++jj) {
                    float e = EC0*kr[0][jj] + EC2*kr[2][jj] + EC3*kr[3][jj]
                            + EC4*kr[4][jj] + EC5*kr[5][jj] + EC6*kr[6][jj];
                    e *= du;
                    float sc = 1e-6f + 1e-3f * fmaxf(fabsf(zr[jj]), fabsf(y5r[jj]));
                    float q = e / sc;
                    psum += q * q;
                }
                for (int o = 16; o >= 1; o >>= 1) psum += __shfl_xor(psum, o, 32);
                float enorm  = sqrtf(psum / (float)H_);
                bool  dn     = (done_s[m] != 0);
                bool  accept = (enorm <= 1.0f) && !dn;
                float factor = fminf(fmaxf(0.9f * powf(enorm + 1e-12f, -0.2f), 0.2f), 10.0f);
                if (accept) {
#pragma unroll
                    for (int jj = 0; jj < 8; ++jj) zr[jj] = y5r[jj];
                }
                if (lane == 0) {
                    if (accept) t_cur[m] = t_cur[m] + du;
                    dtv[m] = dn ? dtv[m] : fmaxf(du * factor, 1e-6f);
                }
            }
            __syncthreads();
        }
        if (tid < MS) t_cur[tid] = t_next;   // interval carry resets t to t_next
        readout(iv);
    }
}

extern "C" void kernel_launch(void* const* d_in, const int* in_sizes, int n_in,
                              void* d_out, int out_size, void* d_ws, size_t ws_size,
                              hipStream_t stream) {
    (void)in_sizes; (void)n_in; (void)out_size; (void)ws_size;
    const float* y_past = (const float*)d_in[0];
    const float* tvec   = (const float*)d_in[1];
    const int*   il     = (const int*)d_in[3];
    const float* wih    = (const float*)d_in[4];
    const float* whh    = (const float*)d_in[5];
    const float* gb     = (const float*)d_in[6];
    const float* gbn    = (const float*)d_in[7];
    const float* w0     = (const float*)d_in[8];
    const float* b0     = (const float*)d_in[9];
    const float* w1     = (const float*)d_in[10];
    const float* b1     = (const float*)d_in[11];
    const float* w2     = (const float*)d_in[12];
    const float* b2     = (const float*)d_in[13];
    const float* row    = (const float*)d_in[14];
    const float* rob    = (const float*)d_in[15];
    _Float16* wpack = (_Float16*)d_ws;
    float*    out   = (float*)d_out;

    pack_weights_kernel<<<256, 256, 0, stream>>>(wih, whh, w0, w1, w2, row, wpack);

    gruode_main<<<NBLK, NTHREADS, SMEM_TOTAL, stream>>>(y_past, tvec, il, gb, gbn,
                                                        b0, b1, b2, rob, wpack, out);
}
